// MultiheadAttention_81913616269553
// MI455X (gfx1250) — matrix operand order repaired and checked by reading, not compiled, not run on hardware
//
#include <hip/hip_runtime.h>
#include <hip/hip_bf16.h>
#include <math.h>

typedef __attribute__((ext_vector_type(16))) _Float16 v16h;
typedef __attribute__((ext_vector_type(8)))  _Float16 v8h;
typedef __attribute__((ext_vector_type(8)))  float    v8f;

#define EMBED 768
#define NHEAD 12
#define HDIM  64
#define BSZ   8
#define SEQ   1024
#define TOKENS (BSZ*SEQ)   // 8192

// ---- WMMA fragment helpers (CDNA5 16-bit layouts, CDNA5 ISA) ----
// A (16x32 f16): lanes 0-15 hold M=lane, K = [kb..kb+7, kb+16..kb+23]
//                lanes16-31 hold M=lane-16, K = [kb+8..kb+15, kb+24..kb+31]
static __device__ __forceinline__ v16h load_a_frag(const _Float16* rowp, int kb, int hi) {
  union { v16h v; v8h h[2]; } u;
  u.h[0] = *(const v8h*)(rowp + kb + hi * 8);
  u.h[1] = *(const v8h*)(rowp + kb + 16 + hi * 8);
  return u.v;
}
// B (32x16 f16): lane j holds column N=j%16; lanes 0-15 K=[kb..kb+7, kb+16..kb+23], lanes 16-31 K=[kb+8..kb+15, kb+24..kb+31].
// colp points at the contiguous K-vector of this lane's column (i.e. a row of W, or a row of Vt).
static __device__ __forceinline__ v16h load_b_frag(const _Float16* colp, int kb, int hi) {
  union { v16h v; v8h h[2]; } u;
  u.h[0] = *(const v8h*)(colp + kb + hi * 8);
  u.h[1] = *(const v8h*)(colp + kb + 16 + hi * 8);
  return u.v;
}
static __device__ __forceinline__ v8f wmma_f16(v16h a, v16h b, v8f c) {
  return __builtin_amdgcn_wmma_f32_16x16x32_f16(false, a, false, b, (short)0, c, false, false);
}

// ---- stage 1: f32 -> f16 conversion ----
__global__ void cvt_f32_to_f16(const float* __restrict__ src, _Float16* __restrict__ dst, long n) {
  long i = (long)blockIdx.x * blockDim.x + threadIdx.x;
  if (i < n) dst[i] = (_Float16)src[i];
}

// ---- stage 2: fused QKV projection ----
// y = x @ W.T + b ; per wave: one 16-token x 16-feature tile of Q, K, V simultaneously.
// Software-pipelined k-loop: next k-step's fragments are in flight while current WMMAs issue.
__global__ void __launch_bounds__(128) qkv_proj(
    const _Float16* __restrict__ X,
    const _Float16* __restrict__ WQ, const _Float16* __restrict__ WK, const _Float16* __restrict__ WV,
    const float* __restrict__ bq, const float* __restrict__ bk, const float* __restrict__ bv,
    _Float16* __restrict__ QH, _Float16* __restrict__ KH, _Float16* __restrict__ VT)
{
  const int lane = threadIdx.x & 31;
  const int lo = lane & 15, hi = lane >> 4;
  const int gw = blockIdx.x * (blockDim.x >> 5) + (threadIdx.x >> 5);
  const int nt = gw % (EMBED / 16);
  const int mt = gw / (EMBED / 16);

  const _Float16* arow = X + (size_t)(mt * 16 + lo) * EMBED;
  const int fcol = nt * 16 + lo;
  const _Float16* wqr = WQ + (size_t)fcol * EMBED;  // row fcol of W == column fcol of W.T
  const _Float16* wkr = WK + (size_t)fcol * EMBED;
  const _Float16* wvr = WV + (size_t)fcol * EMBED;

  v8f cq = {0.f,0.f,0.f,0.f,0.f,0.f,0.f,0.f};
  v8f ck = cq, cv = cq;

  // prologue of the software pipeline
  v16h a0 = load_a_frag(arow, 0, hi);
  v16h q0 = load_b_frag(wqr, 0, hi);
  v16h k0 = load_b_frag(wkr, 0, hi);
  v16h v0 = load_b_frag(wvr, 0, hi);
#pragma unroll 2
  for (int kb = 0; kb < EMBED; kb += 32) {
    const int nk = (kb + 32 < EMBED) ? kb + 32 : 0;  // last iter reloads kb=0 (discarded)
    v16h a1 = load_a_frag(arow, nk, hi);
    v16h q1 = load_b_frag(wqr, nk, hi);
    v16h k1 = load_b_frag(wkr, nk, hi);
    v16h v1 = load_b_frag(wvr, nk, hi);
    cq = wmma_f16(a0, q0, cq);
    ck = wmma_f16(a0, k0, ck);
    cv = wmma_f16(a0, v0, cv);
    a0 = a1; q0 = q1; k0 = k1; v0 = v1;
  }

  const float bqv = bq[fcol], bkv = bk[fcol], bvv = bv[fcol];
  const int hh = fcol >> 6, dd = fcol & 63;
#pragma unroll
  for (int r = 0; r < 8; ++r) {
    const int tok = mt * 16 + r + 8 * hi;          // C/D layout: row = r + 8*(lane>=16)
    const size_t o = (size_t)tok * EMBED + fcol;
    QH[o] = (_Float16)((cq[r] + bqv) * 0.125f);    // fold d^-0.5 into Q
    KH[o] = (_Float16)(ck[r] + bkv);
    const int bb = tok >> 10, n = tok & 1023;
    VT[(((size_t)(bb * NHEAD + hh)) * HDIM + dd) * SEQ + n] = (_Float16)(cv[r] + bvv);
  }
}

// ---- stage 3: fused flash-style attention ----
// One wave = 16 query rows of one (b,h). Online softmax over 32-key chunks; bias streamed once
// with software prefetch of the next chunk; V fragments issued early so the softmax VALU work
// hides their latency; scores/probs never touch HBM.
__global__ void __launch_bounds__(128) attn_fused(
    const float* __restrict__ bias,
    const _Float16* __restrict__ QH, const _Float16* __restrict__ KH,
    const _Float16* __restrict__ VT, _Float16* __restrict__ AH)
{
  __shared__ __align__(64) _Float16 lds[4][16 * 32];
  const int lane = threadIdx.x & 31;
  const int lo = lane & 15, hi = lane >> 4;
  const int wib = threadIdx.x >> 5;
  const int gw = blockIdx.x * (blockDim.x >> 5) + wib;
  const int mb = gw & 63;                 // 16-row block within head
  const int h  = (gw >> 6) % NHEAD;
  const int b  = gw / (64 * NHEAD);
  _Float16* P = lds[wib];

  const _Float16* qrow = QH + ((size_t)(b * SEQ + mb * 16 + lo)) * EMBED + h * HDIM;
  const v16h aq0 = load_a_frag(qrow, 0, hi);   // Q rows, d=0..31
  const v16h aq1 = load_a_frag(qrow, 32, hi);  // Q rows, d=32..63

  const float* brow = bias + ((size_t)(b * NHEAD + h)) * SEQ * SEQ + (size_t)(mb * 16) * SEQ;
  const _Float16* vbase = VT + ((size_t)(b * NHEAD + h)) * HDIM * SEQ;

  float mrun[8], lrun[8];
#pragma unroll
  for (int r = 0; r < 8; ++r) { mrun[r] = -INFINITY; lrun[r] = 0.f; }
  const v8f vz = {0.f,0.f,0.f,0.f,0.f,0.f,0.f,0.f};
  v8f O[4];
#pragma unroll
  for (int t = 0; t < 4; ++t) O[t] = vz;

  for (int kb0 = 0; kb0 < SEQ; kb0 += 32) {
    // ---- issue V-fragment loads first; softmax VALU below hides their latency ----
    v16h vb[4];
#pragma unroll
    for (int t = 0; t < 4; ++t)
      vb[t] = load_b_frag(vbase + (size_t)(t * 16 + lo) * SEQ, kb0, hi);

    // ---- software prefetch of next chunk's 16x32 bias block (global_prefetch_b8) ----
    if (kb0 + 32 < SEQ)
      __builtin_prefetch(brow + (size_t)lo * SEQ + kb0 + 32 + hi * 16, 0, 3);

    // ---- S = Q K^T for 32 keys (two 16x16 tiles), K-dim = d = 64 ----
    v8f S0 = vz, S1 = vz;
    const _Float16* krow0 = KH + ((size_t)(b * SEQ + kb0 + lo)) * EMBED + h * HDIM;
    S0 = wmma_f16(aq0, load_b_frag(krow0, 0, hi), S0);
    S0 = wmma_f16(aq1, load_b_frag(krow0, 32, hi), S0);
    const _Float16* krow1 = krow0 + (size_t)16 * EMBED;
    S1 = wmma_f16(aq0, load_b_frag(krow1, 0, hi), S1);
    S1 = wmma_f16(aq1, load_b_frag(krow1, 32, hi), S1);

    // ---- bias + chunk row-max ----
    float cmax[8];
#pragma unroll
    for (int r = 0; r < 8; ++r) {
      const size_t bo = (size_t)(r + 8 * hi) * SEQ + kb0 + lo;
      S0[r] += brow[bo];
      S1[r] += brow[bo + 16];
      cmax[r] = fmaxf(S0[r], S1[r]);
    }
#pragma unroll
    for (int r = 0; r < 8; ++r)
#pragma unroll
      for (int m = 1; m <= 8; m <<= 1)   // reduce within each 16-lane half (row lives there)
        cmax[r] = fmaxf(cmax[r], __shfl_xor(cmax[r], m, 32));

    // ---- online softmax update ----
    float scale[8], rsum[8];
#pragma unroll
    for (int r = 0; r < 8; ++r) {
      const float nm = fmaxf(mrun[r], cmax[r]);
      scale[r] = __expf(mrun[r] - nm);
      mrun[r] = nm;
      const float p0 = __expf(S0[r] - nm);
      const float p1 = __expf(S1[r] - nm);
      P[(r + 8 * hi) * 32 + lo]      = (_Float16)p0;   // stage P tile in LDS, [row][k]
      P[(r + 8 * hi) * 32 + 16 + lo] = (_Float16)p1;
      rsum[r] = p0 + p1;
    }
#pragma unroll
    for (int r = 0; r < 8; ++r) {
#pragma unroll
      for (int m = 1; m <= 8; m <<= 1)
        rsum[r] += __shfl_xor(rsum[r], m, 32);
      lrun[r] = lrun[r] * scale[r] + rsum[r];
    }
#pragma unroll
    for (int t = 0; t < 4; ++t)
#pragma unroll
      for (int r = 0; r < 8; ++r)
        O[t][r] *= scale[r];

    // per-wave DS ordering: drain stores before re-reading P in A-fragment layout
    asm volatile("s_wait_dscnt 0" ::: "memory");
    const v16h pa = load_a_frag(P + lo * 32, 0, hi);  // P as A (16x32)

    // ---- O += P @ V  (V fragments already in registers) ----
#pragma unroll
    for (int t = 0; t < 4; ++t)
      O[t] = wmma_f16(pa, vb[t], O[t]);
  }

  // ---- normalize and store attn (f16, [token][768]) ----
#pragma unroll
  for (int t = 0; t < 4; ++t)
#pragma unroll
    for (int r = 0; r < 8; ++r) {
      const int tok = b * SEQ + mb * 16 + r + 8 * hi;
      const int f = h * HDIM + t * 16 + lo;
      AH[(size_t)tok * EMBED + f] = (_Float16)(O[t][r] / lrun[r]);
    }
}

// ---- stage 4: output projection, f32 result, software-pipelined k-loop ----
__global__ void __launch_bounds__(128) out_proj(
    const _Float16* __restrict__ A, const _Float16* __restrict__ WO,
    const float* __restrict__ bo, float* __restrict__ out)
{
  const int lane = threadIdx.x & 31;
  const int lo = lane & 15, hi = lane >> 4;
  const int gw = blockIdx.x * (blockDim.x >> 5) + (threadIdx.x >> 5);
  const int nt = gw % (EMBED / 16);
  const int mt = gw / (EMBED / 16);
  const _Float16* arow = A + (size_t)(mt * 16 + lo) * EMBED;
  const int fcol = nt * 16 + lo;
  const _Float16* wrow = WO + (size_t)fcol * EMBED;

  v8f c = {0.f,0.f,0.f,0.f,0.f,0.f,0.f,0.f};
  v16h a0 = load_a_frag(arow, 0, hi);
  v16h w0 = load_b_frag(wrow, 0, hi);
#pragma unroll 4
  for (int kb = 0; kb < EMBED; kb += 32) {
    const int nk = (kb + 32 < EMBED) ? kb + 32 : 0;
    v16h a1 = load_a_frag(arow, nk, hi);
    v16h w1 = load_b_frag(wrow, nk, hi);
    c = wmma_f16(a0, w0, c);
    a0 = a1; w0 = w1;
  }
  const float bb = bo[fcol];
#pragma unroll
  for (int r = 0; r < 8; ++r)
    out[(size_t)(mt * 16 + r + 8 * hi) * EMBED + fcol] = c[r] + bb;
}

extern "C" void kernel_launch(void* const* d_in, const int* in_sizes, int n_in,
                              void* d_out, int out_size, void* d_ws, size_t ws_size,
                              hipStream_t stream)
{
  const float* query = (const float*)d_in[0];
  const float* bias  = (const float*)d_in[1];
  const float* Wq = (const float*)d_in[2];
  const float* bq = (const float*)d_in[3];
  const float* Wk = (const float*)d_in[4];
  const float* bk = (const float*)d_in[5];
  const float* Wv = (const float*)d_in[6];
  const float* bv = (const float*)d_in[7];
  const float* Wo = (const float*)d_in[8];
  const float* bo = (const float*)d_in[9];
  float* out = (float*)d_out;

  const size_t XE = (size_t)TOKENS * EMBED;   // 8192*768
  const size_t WE = (size_t)EMBED * EMBED;    // 768*768
  _Float16* XH  = (_Float16*)d_ws;            // ws layout: ~67.6 MB of f16 scratch
  _Float16* WQH = XH  + XE;
  _Float16* WKH = WQH + WE;
  _Float16* WVH = WKH + WE;
  _Float16* WOH = WVH + WE;
  _Float16* QHp = WOH + WE;
  _Float16* KHp = QHp + XE;
  _Float16* VTp = KHp + XE;
  _Float16* AHp = VTp + XE;

  const int T = 256;
  cvt_f32_to_f16<<<(int)((XE + T - 1) / T), T, 0, stream>>>(query, XH, (long)XE);
  cvt_f32_to_f16<<<(int)((WE + T - 1) / T), T, 0, stream>>>(Wq, WQH, (long)WE);
  cvt_f32_to_f16<<<(int)((WE + T - 1) / T), T, 0, stream>>>(Wk, WKH, (long)WE);
  cvt_f32_to_f16<<<(int)((WE + T - 1) / T), T, 0, stream>>>(Wv, WVH, (long)WE);
  cvt_f32_to_f16<<<(int)((WE + T - 1) / T), T, 0, stream>>>(Wo, WOH, (long)WE);

  // 512 * 48 tiles, 4 waves (128 threads) per block
  qkv_proj<<<(TOKENS / 16) * (EMBED / 16) / 4, 128, 0, stream>>>(
      XH, WQH, WKH, WVH, bq, bk, bv, QHp, KHp, VTp);

  // 8 * 12 * 64 row-block waves, 4 per block
  attn_fused<<<BSZ * NHEAD * (SEQ / 16) / 4, 128, 0, stream>>>(bias, QHp, KHp, VTp, AHp);

  out_proj<<<(TOKENS / 16) * (EMBED / 16) / 4, 128, 0, stream>>>(AHp, WOH, bo, out);
}
